// GCN_13589276525102
// MI455X (gfx1250) — compile-verified, matrix operand order checked by reading, not run on hardware
//
#include <hip/hip_runtime.h>
#include <hip/hip_bf16.h>

// ---------------------------------------------------------------------------
// 2-layer GCN (DGL GraphConv, norm='both') for MI455X / gfx1250.
//   h = relu( Dd^-1/2 A^T Ds^-1/2 (X W1) + b1 )
//   y =       Dd^-1/2 A^T Ds^-1/2 (h W2) + b2
// GEMM runs on v_wmma_f32_16x16x32_bf16 with W pre-packed (bf16) into the
// exact per-lane B-fragment layout so each fragment is one 32B vector load.
// Edge aggregation: wave-per-edge, b128 gather + no-return global f32 atomics.
// ---------------------------------------------------------------------------

typedef __attribute__((ext_vector_type(16))) __bf16 v16bf;
typedef __attribute__((ext_vector_type(8)))  float  v8f;

#define GCN_N 100000
#define GCN_D 128
#define GCN_WPACK_ELEMS (8 * 4 * 32 * 16)   // = 128*128 = 16384

// ---- utility: grid-stride zero fill ---------------------------------------
__global__ void gcn_zero_kernel(float* __restrict__ p, size_t n) {
    size_t i      = (size_t)blockIdx.x * blockDim.x + threadIdx.x;
    size_t stride = (size_t)gridDim.x * blockDim.x;
    for (; i < n; i += stride) p[i] = 0.0f;
}

// ---- degree accumulation (one float atomic per endpoint) ------------------
__global__ void gcn_degree_kernel(const int* __restrict__ src,
                                  const int* __restrict__ dst,
                                  float* __restrict__ deg_src,
                                  float* __restrict__ deg_dst,
                                  int nedges) {
    int i = blockIdx.x * blockDim.x + threadIdx.x;
    if (i < nedges) {
        atomicAdd(&deg_src[src[i]], 1.0f);
        atomicAdd(&deg_dst[dst[i]], 1.0f);
    }
}

// ---- deg -> rsqrt(clip(deg,1)) in place -----------------------------------
__global__ void gcn_norm_kernel(float* __restrict__ p, int n) {
    int i = blockIdx.x * blockDim.x + threadIdx.x;
    if (i < n) p[i] = rsqrtf(fmaxf(p[i], 1.0f));
}

// ---- K(lane, i) mapping for the 16-bit 16x32 WMMA A/B fragment ------------
// ISA layout: lanes 0-15 -> K 0..7 (VGPR0-3) and 16..23 (VGPR4-7);
//             lanes 16-31 -> K 8..15 and 24..31. Element i = VGPR(i/2),half(i%2).
__device__ __forceinline__ int gcn_frag_k(int lane, int i) {
    const int j   = i >> 1;
    const int sub = i & 1;
    const int kk  = (j < 4) ? (2 * j + sub) : (16 + 2 * (j - 4) + sub);
    return kk + 8 * (lane >> 4);
}

// ---- pack W (128x128 f32, row-major) into per-lane bf16 B fragments -------
// Wp[(((nt*4 + kb)*32) + lane)*16 + i] = bf16( W[kb*32 + K(lane,i)][nt*16 + lane%16] )
// 16384 elements total; one thread per element; runs once per layer.
__global__ void gcn_pack_w_kernel(const float* __restrict__ W,
                                  __bf16* __restrict__ Wp) {
    int idx = blockIdx.x * blockDim.x + threadIdx.x;
    if (idx >= GCN_WPACK_ELEMS) return;
    const int i    = idx & 15;
    const int lane = (idx >> 4) & 31;
    const int kb   = (idx >> 9) & 3;
    const int nt   = idx >> 11;
    const int K    = kb * 32 + gcn_frag_k(lane, i);
    const int n    = nt * 16 + (lane & 15);
    Wp[idx] = (__bf16)W[(size_t)K * GCN_D + n];
}

// ---- H = (X @ W) * norm_src[row]  via WMMA bf16 ---------------------------
// Block: 256 threads = 8 waves. Each block owns a 16-row x 128-col tile of H;
// wave w computes the 16x16 sub-tile at columns [16w, 16w+16). K=128 handled
// as 4 steps of v_wmma_f32_16x16x32_bf16. nrows (100000) is a multiple of 16,
// so EXEC is all-1s for every WMMA (ISA requirement). B fragments come from
// the packed bf16 buffer: one aligned 32-byte load per K-step.
__global__ void __launch_bounds__(256)
gcn_gemm_norm_kernel(const float* __restrict__ X,      // [nrows,128] row-major
                     const __bf16* __restrict__ Wp,    // packed fragments
                     const float* __restrict__ norm,   // [nrows]
                     float* __restrict__ H,            // [nrows,128]
                     int nrows) {
    const int lane = threadIdx.x & 31;
    const int wave = threadIdx.x >> 5;          // 0..7 -> column tile
    const int row0 = blockIdx.x * 16;
    const int n0   = wave * 16;
    const int half = lane >> 4;                 // selects K sub-range
    const int m    = lane & 15;                 // row (A) / col (B) in tile

    v8f acc = {};
#pragma unroll
    for (int kb = 0; kb < 4; ++kb) {
        const int k0 = kb * 32;
        // A fragment: two contiguous 8-float runs -> b128 loads + cvt_pk.
        v16bf a;
#pragma unroll
        for (int i = 0; i < 16; ++i) {
            const int K = gcn_frag_k(lane, i);
            a[i] = (__bf16)X[(size_t)(row0 + m) * GCN_D + (k0 + K)];
        }
        // B fragment: pre-packed, 16 contiguous bf16 (32B aligned).
        const v16bf b =
            *(const v16bf*)&Wp[(((size_t)(wave * 4 + kb)) * 32 + lane) * 16];
        acc = __builtin_amdgcn_wmma_f32_16x16x32_bf16(
            /*neg_a=*/false, a, /*neg_b=*/false, b,
            /*c_mod=*/(short)0, acc, /*reuse_a=*/false, /*reuse_b=*/false);
    }

    // C/D 16x16 f32 layout: VGPR i holds row i (lanes 0-15) / row i+8
    // (lanes 16-31), column = lane & 15. Fuse the src-degree row scaling.
#pragma unroll
    for (int i = 0; i < 8; ++i) {
        const int r = row0 + i + 8 * half;
        H[(size_t)r * GCN_D + n0 + m] = acc[i] * norm[r];
    }
}

// ---- edge scatter-add: agg[dst[e]] += H[src[e]] ---------------------------
// One wave per edge; each lane moves one float4 (128 feats / 32 lanes) and
// issues 4 no-return global f32 atomics.
__global__ void __launch_bounds__(256)
gcn_scatter_kernel(const float4* __restrict__ H,   // [nrows, 32] float4 view
                   const int* __restrict__ src,
                   const int* __restrict__ dst,
                   float* __restrict__ agg,        // [nrows,128]
                   int nedges) {
    const int wavesPerBlock = blockDim.x >> 5;
    const int edge = blockIdx.x * wavesPerBlock + (threadIdx.x >> 5);
    const int lane = threadIdx.x & 31;
    if (edge >= nedges) return;
    const int s = src[edge];
    const int d = dst[edge];
    const float4 v = H[(size_t)s * (GCN_D / 4) + lane];
    float* out = agg + (size_t)d * GCN_D + lane * 4;
    atomicAdd(out + 0, v.x);
    atomicAdd(out + 1, v.y);
    atomicAdd(out + 2, v.z);
    atomicAdd(out + 3, v.w);
}

// ---- out = agg * norm_dst[row] + bias[col], optional ReLU -----------------
__global__ void gcn_finalize_kernel(const float* __restrict__ agg,
                                    const float* __restrict__ norm_dst,
                                    const float* __restrict__ bias,
                                    float* __restrict__ out,
                                    int n_elems, int do_relu) {
    int i = blockIdx.x * blockDim.x + threadIdx.x;
    if (i >= n_elems) return;
    const int r = i >> 7;        // /128
    const int c = i & (GCN_D - 1);
    float v = agg[i] * norm_dst[r] + bias[c];
    if (do_relu) v = fmaxf(v, 0.0f);
    out[i] = v;
}

extern "C" void kernel_launch(void* const* d_in, const int* in_sizes, int n_in,
                              void* d_out, int out_size, void* d_ws, size_t ws_size,
                              hipStream_t stream) {
    const float* feat = (const float*)d_in[0];   // [N,128]
    const int*   src  = (const int*)  d_in[1];   // [E]
    const int*   dst  = (const int*)  d_in[2];   // [E]
    const float* W1   = (const float*)d_in[3];   // [128,128]
    const float* b1   = (const float*)d_in[4];   // [128]
    const float* W2   = (const float*)d_in[5];   // [128,128]
    const float* b2   = (const float*)d_in[6];   // [128]
    float* out = (float*)d_out;                  // [N,128]

    const int N = GCN_N;
    const int E = in_sizes[1];
    const size_t NF = (size_t)N * GCN_D;

    // Workspace layout:
    //   normS[N] f32 | normD[N] f32 | Wp1[16384] bf16 | Wp2[16384] bf16 |
    //   bufA[N*128] f32 | bufB[N*128] f32
    // (offsets of Wp1/bufA are 32-byte aligned: 800000 and 865536 bytes)
    float*  normS = (float*)d_ws;
    float*  normD = normS + N;
    __bf16* Wp1   = (__bf16*)(normD + N);
    __bf16* Wp2   = Wp1 + GCN_WPACK_ELEMS;
    float*  bufA  = (float*)(Wp2 + GCN_WPACK_ELEMS);
    float*  bufB  = bufA + NF;

    const int thr = 256;
    const int zgrid = 2048;
    const int egrid = (E + thr - 1) / thr;
    const int ngrid2 = (2 * N + thr - 1) / thr;
    const int pgrid = GCN_WPACK_ELEMS / thr;              // 64
    const int gemm_grid = N / 16;                         // 6250, exact
    const int scat_grid = (E + (thr / 32) - 1) / (thr / 32);
    const int fin_grid  = (int)((NF + thr - 1) / thr);

    // --- degree norms + packed weights (shared setup) ---
    gcn_zero_kernel<<<zgrid, thr, 0, stream>>>(normS, (size_t)2 * N);
    gcn_degree_kernel<<<egrid, thr, 0, stream>>>(src, dst, normS, normD, E);
    gcn_norm_kernel<<<ngrid2, thr, 0, stream>>>(normS, 2 * N);
    gcn_pack_w_kernel<<<pgrid, thr, 0, stream>>>(W1, Wp1);
    gcn_pack_w_kernel<<<pgrid, thr, 0, stream>>>(W2, Wp2);

    // --- layer 1 ---
    gcn_gemm_norm_kernel<<<gemm_grid, thr, 0, stream>>>(feat, Wp1, normS, bufA, N);
    gcn_zero_kernel<<<zgrid, thr, 0, stream>>>(bufB, NF);
    gcn_scatter_kernel<<<scat_grid, thr, 0, stream>>>((const float4*)bufA, src, dst, bufB, E);
    gcn_finalize_kernel<<<fin_grid, thr, 0, stream>>>(bufB, normD, b1, bufA, (int)NF, 1);

    // --- layer 2 (scatter straight into d_out) ---
    gcn_gemm_norm_kernel<<<gemm_grid, thr, 0, stream>>>(bufA, Wp2, normS, bufB, N);
    gcn_zero_kernel<<<zgrid, thr, 0, stream>>>(out, NF);
    gcn_scatter_kernel<<<scat_grid, thr, 0, stream>>>((const float4*)bufB, src, dst, out, E);
    gcn_finalize_kernel<<<fin_grid, thr, 0, stream>>>(out, normD, b2, out, (int)NF, 0);
}
